// GPTLanguageModel_48473000902971
// MI455X (gfx1250) — matrix operand order repaired and checked by reading, not compiled, not run on hardware
//
#include <hip/hip_runtime.h>
#include <hip/hip_bf16.h>
#include <math.h>

// ---------------- problem constants ----------------
#define BB   16
#define TT   256
#define CC   384
#define HH   6
#define DD   64
#define LLY  6
#define FFD  1536
#define VV   32000
#define RR   (BB*TT)          // 4096 rows of the residual stream

typedef __attribute__((ext_vector_type(16))) _Float16 v16h;
typedef __attribute__((ext_vector_type(8)))  float    v8f;
typedef int v4i_vs __attribute__((vector_size(16)));   // matches builtin param type

// ---- gfx1250 async global->LDS copy (ASYNCcnt path), with safe fallbacks ----
#if defined(__HIP_DEVICE_COMPILE__) && __has_builtin(__builtin_amdgcn_global_load_async_to_lds_b128)
#define USE_ASYNC_LDS 1
#endif

__device__ __forceinline__ void cp_async16(const _Float16* g, _Float16* l)
{
#ifdef USE_ASYNC_LDS
    __builtin_amdgcn_global_load_async_to_lds_b128(
        (__attribute__((address_space(1))) v4i_vs*)(void*)const_cast<_Float16*>(g),
        (__attribute__((address_space(3))) v4i_vs*)(void*)l,
        0, 0);
#else
    *(float4*)l = *(const float4*)g;
#endif
}

__device__ __forceinline__ void wait_async()
{
#ifdef USE_ASYNC_LDS
#if __has_builtin(__builtin_amdgcn_s_wait_asynccnt)
    __builtin_amdgcn_s_wait_asynccnt(0);
#else
    asm volatile("s_wait_asynccnt 0" ::: "memory");
#endif
#endif
}

// ---------------- embedding: x = tok_emb[ids] + pos_emb ----------------
__global__ void embed_kernel(const int* __restrict__ x_ids,
                             const float* __restrict__ tok_emb,
                             const float* __restrict__ pos_emb,
                             float* __restrict__ x)
{
    int r = blockIdx.x;            // 0..RR-1
    int c = threadIdx.x;           // 0..CC-1 (blockDim = 384)
    int t = r % TT;
    int id = x_ids[r];
    x[(size_t)r * CC + c] = tok_emb[(size_t)id * CC + c] + pos_emb[(size_t)t * CC + c];
}

// -------- weight convert fp32 [K,N] (with optional head-split) -> f16 [N,K] --------
// src element for (n,k):  (n/Dsplit)*(K*Dsplit) + k*Dsplit + (n%Dsplit)
// Dsplit==N  -> plain row-major [K,N];  Dsplit==DD -> [H][K][D] per-head weights.
__global__ void convertT_kernel(const float* __restrict__ in, _Float16* __restrict__ out,
                                int K, int N, int Dsplit)
{
    int idx = blockIdx.x * 256 + threadIdx.x;
    if (idx >= K * N) return;
    int n = idx / K;
    int k = idx - n * K;
    size_t src = (size_t)(n / Dsplit) * ((size_t)K * Dsplit) + (size_t)k * Dsplit + (n % Dsplit);
    out[(size_t)n * K + k] = (_Float16)in[src];
}

// ---------------- layernorm (fp32 in, f16 out), one row per block ----------------
__global__ void ln_kernel(const float* __restrict__ x,
                          const float* __restrict__ gamma,
                          const float* __restrict__ beta,
                          _Float16* __restrict__ out)
{
    __shared__ float red[128];
    int r = blockIdx.x, t = threadIdx.x;     // blockDim = 128
    const float* xr = x + (size_t)r * CC;

    float s = 0.f;
    for (int c = t; c < CC; c += 128) s += xr[c];
    red[t] = s; __syncthreads();
    for (int off = 64; off > 0; off >>= 1) { if (t < off) red[t] += red[t + off]; __syncthreads(); }
    float mean = red[0] / (float)CC; __syncthreads();

    float v = 0.f;
    for (int c = t; c < CC; c += 128) { float d = xr[c] - mean; v += d * d; }
    red[t] = v; __syncthreads();
    for (int off = 64; off > 0; off >>= 1) { if (t < off) red[t] += red[t + off]; __syncthreads(); }
    float inv = rsqrtf(red[0] / (float)CC + 1e-5f);

    for (int c = t; c < CC; c += 128)
        out[(size_t)r * CC + c] = (_Float16)((xr[c] - mean) * inv * gamma[c] + beta[c]);
}

// ---------------- WMMA GEMM with LDS staging ----------------
// out[M,N] = A[M,K](f16) @ Bt[N,K](f16)^T (+bias)(+resid)(relu)
// block = (32,4): 64x64 output tile per block; wave w owns the 64x16 strip at
// columns [w*16, w*16+16). A/B k-slabs (64x32 halves each) staged in LDS via
// async global->LDS copies; one B fragment is reused across 4 WMMAs.
// Requires M%64==0, N%64==0, K%32==0 (true for all GEMMs here).
__global__ void __launch_bounds__(128)
gemm_wmma_f16(const _Float16* __restrict__ A, const _Float16* __restrict__ Bt,
              const float* __restrict__ bias, const float* __restrict__ resid,
              float* __restrict__ outF, _Float16* __restrict__ outH,
              int M, int N, int K, int relu)
{
    __shared__ __align__(16) _Float16 sA[64 * 32];
    __shared__ __align__(16) _Float16 sB[64 * 32];

    const int lane = threadIdx.x;              // 0..31
    const int wv   = threadIdx.y;              // 0..3 : N sub-tile
    const int tid  = wv * 32 + lane;           // 0..127
    const int n0   = blockIdx.x * 64;
    const int m0   = blockIdx.y * 64;

    // fragment geometry (per lane): contiguous k order, same for A and B
    const int lrow = lane & 15;
    const int akb  = (lane >> 4) << 4;         // A: K halves 0/16, 16 contiguous
    const int bkb  = (lane >> 4) << 4;         // B: K halves 0/16, 16 contiguous

    // cooperative copy geometry: tile = 64 rows x 32 halves = 256 x 16B chunks;
    // each thread moves 2 chunks per tile (rows tid/4 and tid/4+32).
    const int crow = tid >> 2;                 // 0..31
    const int cc4  = (tid & 3) * 8;            // half offset within row

    v8f acc[4] = {};

    for (int k0 = 0; k0 < K; k0 += 32) {
        __syncthreads();   // previous iteration's LDS reads complete
        cp_async16(A  + (size_t)(m0 + crow)      * K + k0 + cc4, sA + (crow)      * 32 + cc4);
        cp_async16(A  + (size_t)(m0 + crow + 32) * K + k0 + cc4, sA + (crow + 32) * 32 + cc4);
        cp_async16(Bt + (size_t)(n0 + crow)      * K + k0 + cc4, sB + (crow)      * 32 + cc4);
        cp_async16(Bt + (size_t)(n0 + crow + 32) * K + k0 + cc4, sB + (crow + 32) * 32 + cc4);
        wait_async();
        __syncthreads();   // LDS slab visible to all waves

        v16h b;
        const _Float16* bsrc = sB + (wv * 16 + lrow) * 32 + bkb;
#pragma unroll
        for (int e = 0; e < 16; ++e) b[e] = bsrc[e];

#pragma unroll
        for (int mi = 0; mi < 4; ++mi) {
            v16h a;
            const _Float16* asrc = sA + (mi * 16 + lrow) * 32 + akb;
#pragma unroll
            for (int e = 0; e < 16; ++e) a[e] = asrc[e];
            acc[mi] = __builtin_amdgcn_wmma_f32_16x16x32_f16(false, a, false, b,
                                                             (short)0, acc[mi], false, false);
        }
    }

    // C/D layout: VGPR j -> row j (lanes 0-15) / row j+8 (lanes 16-31), col = lane%16
    const int ccol  = n0 + wv * 16 + lrow;
    const int rbase = (lane >> 4) << 3;
#pragma unroll
    for (int mi = 0; mi < 4; ++mi) {
#pragma unroll
        for (int j = 0; j < 8; ++j) {
            int r = m0 + mi * 16 + rbase + j;
            size_t idx = (size_t)r * N + ccol;
            float val = acc[mi][j];
            if (bias)  val += bias[ccol];
            if (resid) val += resid[idx];
            if (relu)  val = val > 0.f ? val : 0.f;
            if (outF)  outF[idx] = val;
            if (outH)  outH[idx] = (_Float16)val;
        }
    }
}

// ---------------- causal attention, fp32 softmax; one (b,h,t) row per block ----------------
__global__ void attn_kernel(const _Float16* __restrict__ q, const _Float16* __restrict__ k,
                            const _Float16* __restrict__ v, _Float16* __restrict__ o,
                            float scale)
{
    int t = blockIdx.x, h = blockIdx.y, b = blockIdx.z;
    int s = threadIdx.x;                              // 0..255
    __shared__ float att[TT];
    __shared__ float red[TT];

    const _Float16* qp = q + ((size_t)(b * TT + t)) * CC + h * DD;
    float sc = -1e30f;
    if (s <= t) {
        const _Float16* kp = k + ((size_t)(b * TT + s)) * CC + h * DD;
        float d = 0.f;
#pragma unroll
        for (int i = 0; i < DD; ++i) d += (float)qp[i] * (float)kp[i];
        sc = d * scale;
    }
    red[s] = sc; __syncthreads();
    for (int off = 128; off > 0; off >>= 1) { if (s < off) red[s] = fmaxf(red[s], red[s + off]); __syncthreads(); }
    float mx = red[0]; __syncthreads();

    float p = (s <= t) ? expf(sc - mx) : 0.f;
    att[s] = p;
    red[s] = p; __syncthreads();
    for (int off = 128; off > 0; off >>= 1) { if (s < off) red[s] += red[s + off]; __syncthreads(); }
    float denom = red[0]; __syncthreads();

    if (s < DD) {
        float accv = 0.f;
        for (int j = 0; j <= t; ++j)
            accv += att[j] * (float)v[((size_t)(b * TT + j)) * CC + h * DD + s];
        o[((size_t)(b * TT + t)) * CC + h * DD + s] = (_Float16)(accv / denom);
    }
}

// ---------------- loss: per-row log-softmax NLL, then mean ----------------
__global__ void loss_rows_kernel(const float* __restrict__ logits, const int* __restrict__ y_ids,
                                 float* __restrict__ row_loss)
{
    __shared__ float red[256];
    int r = blockIdx.x, t = threadIdx.x;
    const float* Lr = logits + (size_t)r * VV;

    float mx = -1e30f;
    for (int w = t; w < VV; w += 256) mx = fmaxf(mx, Lr[w]);
    red[t] = mx; __syncthreads();
    for (int off = 128; off > 0; off >>= 1) { if (t < off) red[t] = fmaxf(red[t], red[t + off]); __syncthreads(); }
    mx = red[0]; __syncthreads();

    float s = 0.f;
    for (int w = t; w < VV; w += 256) s += expf(Lr[w] - mx);
    red[t] = s; __syncthreads();
    for (int off = 128; off > 0; off >>= 1) { if (t < off) red[t] += red[t + off]; __syncthreads(); }
    if (t == 0) row_loss[r] = mx + logf(red[0]) - Lr[y_ids[r]];
}

__global__ void loss_reduce_kernel(const float* __restrict__ row_loss, float* __restrict__ out)
{
    __shared__ float red[256];
    int t = threadIdx.x;
    float s = 0.f;
    for (int i = t; i < RR; i += 256) s += row_loss[i];
    red[t] = s; __syncthreads();
    for (int off = 128; off > 0; off >>= 1) { if (t < off) red[t] += red[t + off]; __syncthreads(); }
    if (t == 0) out[0] = red[0] / (float)RR;
}

// ---------------- orchestration ----------------
extern "C" void kernel_launch(void* const* d_in, const int* in_sizes, int n_in,
                              void* d_out, int out_size, void* d_ws, size_t ws_size,
                              hipStream_t stream)
{
    (void)in_sizes; (void)n_in; (void)out_size; (void)ws_size;

    const int*   x_ids   = (const int*)  d_in[0];
    const int*   y_ids   = (const int*)  d_in[1];
    const float* tok_emb = (const float*)d_in[2];
    const float* pos_emb = (const float*)d_in[3];
    const float* Wq      = (const float*)d_in[4];
    const float* Wk      = (const float*)d_in[5];
    const float* Wv      = (const float*)d_in[6];
    const float* Wo      = (const float*)d_in[7];
    const float* bo      = (const float*)d_in[8];
    const float* ln1_s   = (const float*)d_in[9];
    const float* ln1_b   = (const float*)d_in[10];
    const float* ln2_s   = (const float*)d_in[11];
    const float* ln2_b   = (const float*)d_in[12];
    const float* W1      = (const float*)d_in[13];
    const float* b1      = (const float*)d_in[14];
    const float* W2      = (const float*)d_in[15];
    const float* b2      = (const float*)d_in[16];
    const float* lnf_s   = (const float*)d_in[17];
    const float* lnf_b   = (const float*)d_in[18];
    const float* Whead   = (const float*)d_in[19];
    const float* bhead   = (const float*)d_in[20];

    float* logits = (float*)d_out;                       // [RR, VV]
    float* loss   = logits + (size_t)RR * VV;            // scalar at the end

    // ---- workspace bump allocator ----
    char* ws = (char*)d_ws;
    size_t off = 0;
    auto alloc = [&](size_t bytes) -> void* {
        off = (off + 255) & ~(size_t)255;
        void* p = ws + off;
        off += bytes;
        return p;
    };
    float*     x    = (float*)    alloc((size_t)RR * CC * 4);   // residual stream fp32
    _Float16*  h    = (_Float16*) alloc((size_t)RR * CC * 2);   // LN output f16
    _Float16*  qb   = (_Float16*) alloc((size_t)RR * CC * 2);
    _Float16*  kb   = (_Float16*) alloc((size_t)RR * CC * 2);
    _Float16*  vb   = (_Float16*) alloc((size_t)RR * CC * 2);
    _Float16*  ob   = (_Float16*) alloc((size_t)RR * CC * 2);
    _Float16*  f1   = (_Float16*) alloc((size_t)RR * FFD * 2);
    _Float16*  wqT  = (_Float16*) alloc((size_t)CC * CC * 2);   // [N=H*D, K=C]
    _Float16*  wkT  = (_Float16*) alloc((size_t)CC * CC * 2);
    _Float16*  wvT  = (_Float16*) alloc((size_t)CC * CC * 2);
    _Float16*  woT  = (_Float16*) alloc((size_t)CC * CC * 2);   // [N=C, K=H*D]
    _Float16*  w1T  = (_Float16*) alloc((size_t)FFD * CC * 2);  // [N=FF, K=C]
    _Float16*  w2T  = (_Float16*) alloc((size_t)CC * FFD * 2);  // [N=C, K=FF]
    _Float16*  whT  = (_Float16*) alloc((size_t)VV * CC * 2);   // [N=V, K=C]
    float*     rlos = (float*)    alloc((size_t)RR * 4);

    const float scale = 1.0f / sqrtf((float)CC);   // reference scales by C^-0.5

    dim3 gblk(32, 4);
    auto ggrid = [](int M, int N) { return dim3((unsigned)(N / 64), (unsigned)(M / 64)); };
    auto cgrid = [](int elems)    { return dim3((unsigned)((elems + 255) / 256)); };

    // embeddings
    embed_kernel<<<RR, CC, 0, stream>>>(x_ids, tok_emb, pos_emb, x);

    for (int l = 0; l < LLY; ++l) {
        // weight conversion (fp32 -> transposed f16)
        convertT_kernel<<<cgrid(CC * CC), 256, 0, stream>>>(Wq + (size_t)l * HH * CC * DD, wqT, CC, CC, DD);
        convertT_kernel<<<cgrid(CC * CC), 256, 0, stream>>>(Wk + (size_t)l * HH * CC * DD, wkT, CC, CC, DD);
        convertT_kernel<<<cgrid(CC * CC), 256, 0, stream>>>(Wv + (size_t)l * HH * CC * DD, wvT, CC, CC, DD);
        convertT_kernel<<<cgrid(CC * CC), 256, 0, stream>>>(Wo + (size_t)l * CC * CC,      woT, CC, CC, CC);
        convertT_kernel<<<cgrid(CC * FFD), 256, 0, stream>>>(W1 + (size_t)l * CC * FFD,    w1T, CC,  FFD, FFD);
        convertT_kernel<<<cgrid(FFD * CC), 256, 0, stream>>>(W2 + (size_t)l * FFD * CC,    w2T, FFD, CC,  CC);

        // h = LN1(x)
        ln_kernel<<<RR, 128, 0, stream>>>(x, ln1_s + (size_t)l * CC, ln1_b + (size_t)l * CC, h);

        // q,k,v = h @ W{q,k,v}   (all heads fused along N)
        gemm_wmma_f16<<<ggrid(RR, CC), gblk, 0, stream>>>(h, wqT, nullptr, nullptr, nullptr, qb, RR, CC, CC, 0);
        gemm_wmma_f16<<<ggrid(RR, CC), gblk, 0, stream>>>(h, wkT, nullptr, nullptr, nullptr, kb, RR, CC, CC, 0);
        gemm_wmma_f16<<<ggrid(RR, CC), gblk, 0, stream>>>(h, wvT, nullptr, nullptr, nullptr, vb, RR, CC, CC, 0);

        // causal attention
        attn_kernel<<<dim3(TT, HH, BB), 256, 0, stream>>>(qb, kb, vb, ob, scale);

        // x = x + o @ Wo + bo
        gemm_wmma_f16<<<ggrid(RR, CC), gblk, 0, stream>>>(ob, woT, bo + (size_t)l * CC, x, x, nullptr, RR, CC, CC, 0);

        // h = LN2(x)
        ln_kernel<<<RR, 128, 0, stream>>>(x, ln2_s + (size_t)l * CC, ln2_b + (size_t)l * CC, h);

        // f1 = relu(h @ W1 + b1) ;  x = x + f1 @ W2 + b2
        gemm_wmma_f16<<<ggrid(RR, FFD), gblk, 0, stream>>>(h,  w1T, b1 + (size_t)l * FFD, nullptr, nullptr, f1, RR, FFD, CC, 1);
        gemm_wmma_f16<<<ggrid(RR, CC),  gblk, 0, stream>>>(f1, w2T, b2 + (size_t)l * CC,  x,       x, nullptr, RR, CC, FFD, 0);
    }

    // final LN + head GEMM into d_out
    ln_kernel<<<RR, 128, 0, stream>>>(x, lnf_s, lnf_b, h);
    convertT_kernel<<<cgrid(CC * VV), 256, 0, stream>>>(Whead, whT, CC, VV, VV);
    gemm_wmma_f16<<<ggrid(RR, VV), gblk, 0, stream>>>(h, whT, bhead, nullptr, logits, nullptr, RR, VV, CC, 0);

    // loss
    loss_rows_kernel<<<RR, 256, 0, stream>>>(logits, y_ids, rlos);
    loss_reduce_kernel<<<1, 256, 0, stream>>>(rlos, loss);
}
